// EfficientAttention_43026982371360
// MI455X (gfx1250) — matrix operand order repaired and checked by reading, not compiled, not run on hardware
//
#include <hip/hip_runtime.h>
#include <hip/hip_bf16.h>

typedef __bf16 bf16;
typedef __attribute__((ext_vector_type(16))) __bf16 v16bf;
typedef __attribute__((ext_vector_type(8)))  __bf16 v8bf;
typedef __attribute__((ext_vector_type(8)))  float  v8f;
typedef __attribute__((ext_vector_type(4)))  unsigned int v4u;
typedef __attribute__((ext_vector_type(4)))  int v4i;
typedef __attribute__((ext_vector_type(8)))  int v8i;

#define N_TOK 65536
#define C_DIM 1024
#define K_DIM 1024
#define H_CNT 16
#define DH    64
#define QKV   3072

// ---------------- helpers ----------------

__device__ __forceinline__ v16bf load_frag(const bf16* lo, const bf16* hi) {
    v8bf l = *reinterpret_cast<const v8bf*>(lo);
    v8bf h = *reinterpret_cast<const v8bf*>(hi);
    return __builtin_shufflevector(l, h, 0,1,2,3,4,5,6,7,8,9,10,11,12,13,14,15);
}

__device__ __forceinline__ v8f wmma_bf16(v16bf a, v16bf b, v8f c) {
    // 8 args: (neg_a, A, neg_b, B, c_mod, C, reuse_a, reuse_b)
    return __builtin_amdgcn_wmma_f32_16x16x32_bf16(false, a, false, b, (short)0, c, false, false);
}

__device__ __forceinline__ unsigned enc_f32(float f) {
    unsigned b = __float_as_uint(f);
    return (b & 0x80000000u) ? ~b : (b | 0x80000000u);
}
__device__ __forceinline__ float dec_f32(unsigned u) {
    unsigned b = (u & 0x80000000u) ? (u & 0x7FFFFFFFu) : ~u;
    return __uint_as_float(b);
}

// Low 32 bits of a generic pointer to __shared__ = LDS byte offset (ISA 10.2 aperture rule)
__device__ __forceinline__ unsigned lds_off(const void* p) {
    return (unsigned)(unsigned long long)p;
}

// ---- Tensor Data Mover: 2D bf16 tile global->LDS with LDS row padding ----
// D# per CDNA5 ISA ch.8: group0 {count,lds_addr,global_addr,type=2},
// group1 {data_size=2B, pad_enable, pad_interval, pad_amount, tensor dims, tile dims, dim0 stride}
__device__ __forceinline__ void tdm_load_2d_bf16(unsigned lds_byte_addr, const void* gaddr,
                                                 unsigned tile_w, unsigned tile_h,
                                                 unsigned row_stride_elems,
                                                 unsigned tensor_w, unsigned tensor_h,
                                                 unsigned pad_interval_enc, unsigned pad_amount_enc) {
    unsigned long long ga = (unsigned long long)(size_t)gaddr;
    v4u g0;
    g0[0] = 1u;                                                  // count=1 (valid), user mode
    g0[1] = lds_byte_addr;                                       // lds_addr
    g0[2] = (unsigned)(ga & 0xFFFFFFFFull);                      // global_addr[31:0]
    g0[3] = (unsigned)((ga >> 32) & 0x1FFFFFFull) | (2u << 30);  // global_addr[56:32], type=2
    v8i g1;
    g1[0] = (int)((1u << 16)                       // data_size = 1 -> 2 bytes
                | (1u << 20)                       // pad_enable
                | (pad_interval_enc << 22)
                | (pad_amount_enc << 25));
    g1[1] = (int)((tensor_w & 0xFFFFu) << 16);                         // tensor_dim0[15:0]
    g1[2] = (int)(((tensor_w >> 16) & 0xFFFFu) | ((tensor_h & 0xFFFFu) << 16));
    g1[3] = (int)(((tensor_h >> 16) & 0xFFFFu) | ((tile_w & 0xFFFFu) << 16)); // tile_dim0
    g1[4] = (int)(tile_h & 0xFFFFu);                                   // tile_dim1, tile_dim2=0
    g1[5] = (int)row_stride_elems;                                     // tensor_dim0_stride[31:0]
    g1[6] = 0;
    g1[7] = 0;
    v4i gz = {0, 0, 0, 0};
#if __clang_major__ >= 23
    v8i gz8 = {0, 0, 0, 0, 0, 0, 0, 0};
    __builtin_amdgcn_tensor_load_to_lds(g0, g1, gz, gz, gz8, 0);
#else
    __builtin_amdgcn_tensor_load_to_lds(g0, g1, gz, gz, 0);
#endif
}

// ---------------- conversion / packing ----------------

__global__ void cvt_f32_bf16_kernel(const float* __restrict__ src, bf16* __restrict__ dst, int count) {
    for (int i = blockIdx.x * blockDim.x + threadIdx.x; i < count; i += gridDim.x * blockDim.x)
        dst[i] = (bf16)src[i];
}

__global__ void packw_kernel(const float* __restrict__ Wk, const float* __restrict__ Wq,
                             const float* __restrict__ Wv, bf16* __restrict__ W) {
    const int total = C_DIM * QKV;
    for (int i = blockIdx.x * blockDim.x + threadIdx.x; i < total; i += gridDim.x * blockDim.x) {
        int c = i / QKV;
        int col = i - c * QKV;
        int sel = col >> 10;          // 0:Wk 1:Wq 2:Wv
        int j = col & 1023;
        const float* src = (sel == 0) ? Wk : (sel == 1) ? Wq : Wv;
        W[i] = (bf16)src[c * K_DIM + j];
    }
}

__global__ void packb_kernel(const float* __restrict__ bk, const float* __restrict__ bq,
                             const float* __restrict__ bv, float* __restrict__ b) {
    int i = blockIdx.x * blockDim.x + threadIdx.x;
    if (i < QKV) {
        int sel = i >> 10;
        const float* src = (sel == 0) ? bk : (sel == 1) ? bq : bv;
        b[i] = src[i & 1023];
    }
}

__global__ void init_stats_kernel(unsigned* __restrict__ colmax_u, float* __restrict__ colsum,
                                  float* __restrict__ ctx) {
    for (int i = blockIdx.x * blockDim.x + threadIdx.x; i < H_CNT * DH * DH; i += gridDim.x * blockDim.x) {
        ctx[i] = 0.0f;
        if (i < K_DIM) { colmax_u[i] = 0x007FFFFFu; /* enc(-inf) */ colsum[i] = 0.0f; }
    }
}

// ---------------- generic bf16 WMMA GEMM  C[M,N] = A[M,K]*B[K,N] + bias ----------------
// BM=128 BN=128 BK=32, 256 threads = 8 waves, each wave: 32 rows x 64 cols (2x4 tiles)
// A tile staged by the Tensor Data Mover (with LDS padding); B staged transposed manually.

#define BM 128
#define BN 128
#define BK 32
#define LDPAD 40   // padded LDS stride in elements (80B: 16B-aligned frags, conflict-reducing)

template<bool OUT_BF16>
__global__ __launch_bounds__(256)
void gemm_bf16_kernel(const bf16* __restrict__ A, const bf16* __restrict__ B,
                      const float* __restrict__ bias, void* __restrict__ Cout,
                      int M, int Nc, int Kc) {
    __shared__ alignas(16) bf16 alds[2][BM * LDPAD];
    __shared__ alignas(16) bf16 blds[2][BN * LDPAD];

    const int tid  = threadIdx.x;
    const int lane = tid & 31, wid = tid >> 5;
    const int lm = lane & 15, lh = lane >> 4;
    const int wr = wid & 3;       // 4 row groups of 32
    const int wc = wid >> 2;      // 2 col groups of 64
    const int m0 = blockIdx.y * BM;
    const int n0 = blockIdx.x * BN;

    v8f acc[2][4];
    #pragma unroll
    for (int i = 0; i < 2; ++i)
        #pragma unroll
        for (int c = 0; c < 4; ++c)
            #pragma unroll
            for (int e = 0; e < 8; ++e) acc[i][c][e] = 0.0f;

    // TDM: A tile 128x32 bf16, rows 64B + 16B pad -> LDS stride 80B (LDPAD=40)
    auto issue_a = [&](int kt, int buf) {
        if (wid == 0)
            tdm_load_2d_bf16(lds_off(&alds[buf][0]),
                             A + (size_t)m0 * Kc + (size_t)kt * BK,
                             BK, BM, (unsigned)Kc, (unsigned)Kc, (unsigned)M,
                             3u /*16 DW = 64B*/, 3u /*4 DW = 16B*/);
    };
    // B tile: 32x128, stored transposed blds[n][k]
    auto load_b = [&](int kt, int buf) {
        const bf16* Bg = B + (size_t)kt * BK * Nc + n0;
        __builtin_prefetch(Bg + (size_t)BK * Nc, 0, 0);
        for (int c = tid; c < BN * BK; c += 256) {
            int nl = c & 127, kl = c >> 7;
            blds[buf][nl * LDPAD + kl] = Bg[(size_t)kl * Nc + nl];
        }
    };

    const int nk = Kc / BK;
    issue_a(0, 0);
    load_b(0, 0);
    if (wid == 0) __builtin_amdgcn_s_wait_tensorcnt(0);
    __syncthreads();

    for (int kt = 0; kt < nk; ++kt) {
        const int buf = kt & 1;
        if (kt + 1 < nk) {
            issue_a(kt + 1, buf ^ 1);
            load_b(kt + 1, buf ^ 1);
        }
        const bf16* al = &alds[buf][0];
        const bf16* bl = &blds[buf][0];

        v16bf afr[2], bfr[4];
        #pragma unroll
        for (int i = 0; i < 2; ++i) {
            const bf16* p = al + (wr * 32 + i * 16 + lm) * LDPAD + lh * 8;
            afr[i] = load_frag(p, p + 16);
        }
        #pragma unroll
        for (int c = 0; c < 4; ++c) {
            const bf16* p = bl + (wc * 64 + c * 16 + lm) * LDPAD + lh * 8;
            bfr[c] = load_frag(p, p + 16);
        }
        #pragma unroll
        for (int i = 0; i < 2; ++i)
            #pragma unroll
            for (int c = 0; c < 4; ++c)
                acc[i][c] = wmma_bf16(afr[i], bfr[c], acc[i][c]);

        if (wid == 0 && kt + 1 < nk) __builtin_amdgcn_s_wait_tensorcnt(0);
        __syncthreads();
    }

    // epilogue: C/D layout — VGPR r: lanes 0-15 M=r, lanes 16-31 M=r+8; N=lane&15
    #pragma unroll
    for (int i = 0; i < 2; ++i) {
        #pragma unroll
        for (int c = 0; c < 4; ++c) {
            const int nn = n0 + wc * 64 + c * 16 + lm;
            const float bvv = bias ? bias[nn] : 0.0f;
            #pragma unroll
            for (int r = 0; r < 8; ++r) {
                const int row = m0 + wr * 32 + i * 16 + r + lh * 8;
                const float val = acc[i][c][r] + bvv;
                if (OUT_BF16) ((bf16*)Cout)[(size_t)row * Nc + nn] = (bf16)val;
                else          ((float*)Cout)[(size_t)row * Nc + nn] = val;
            }
        }
    }
}

// ---------------- key column softmax stats (over all N tokens) ----------------

__global__ __launch_bounds__(256) void colmax_kernel(const bf16* __restrict__ kqv,
                                                     unsigned* __restrict__ colmax_u) {
    const int col = blockIdx.x * 256 + threadIdx.x;       // 1024 columns
    const int row0 = blockIdx.y * 512;
    float m = -INFINITY;
    for (int r = 0; r < 512; ++r)
        m = fmaxf(m, (float)kqv[(size_t)(row0 + r) * QKV + col]);
    atomicMax(&colmax_u[col], enc_f32(m));
}

__global__ __launch_bounds__(256) void colsum_kernel(const bf16* __restrict__ kqv,
                                                     const unsigned* __restrict__ colmax_u,
                                                     float* __restrict__ colsum) {
    const int col = blockIdx.x * 256 + threadIdx.x;
    const int row0 = blockIdx.y * 512;
    const float mx = dec_f32(colmax_u[col]);
    float s = 0.0f;
    for (int r = 0; r < 512; ++r)
        s += __expf((float)kqv[(size_t)(row0 + r) * QKV + col] - mx);
    atomicAdd(&colsum[col], s);
}

// ---------------- query softmax over 64 head features (one wave per (n,h)) ----------------

__global__ __launch_bounds__(256) void qsoftmax_kernel(bf16* __restrict__ kqv) {
    const int gw = (blockIdx.x * 256 + threadIdx.x) >> 5;
    const int lane = threadIdx.x & 31;
    const int n = gw >> 4;
    const int h = gw & 15;
    bf16* row = kqv + (size_t)n * QKV + K_DIM + h * DH;
    float x0 = (float)row[lane];
    float x1 = (float)row[lane + 32];
    float m = fmaxf(x0, x1);
    #pragma unroll
    for (int o = 16; o > 0; o >>= 1) m = fmaxf(m, __shfl_xor(m, o, 32));
    const float e0 = __expf(x0 - m), e1 = __expf(x1 - m);
    float s = e0 + e1;
    #pragma unroll
    for (int o = 16; o > 0; o >>= 1) s += __shfl_xor(s, o, 32);
    const float inv = 1.0f / s;
    row[lane]      = (bf16)(e0 * inv);
    row[lane + 32] = (bf16)(e1 * inv);
}

// ---------------- ctx[h,kc,vc] = sum_n softmax_k(keys)[n,kc] * v[n,vc] ----------------
// block = (head, N-slice of 1024 tokens); stages k^T and v^T tiles in LDS, WMMA accumulate

#define NT  128
#define NTP 136

__global__ __launch_bounds__(256) void ctx_kernel(const bf16* __restrict__ kqv,
                                                  const unsigned* __restrict__ colmax_u,
                                                  const float* __restrict__ colsum,
                                                  float* __restrict__ ctx) {
    __shared__ alignas(16) bf16 klds[DH * NTP];   // [kc][n]
    __shared__ alignas(16) bf16 vlds[DH * NTP];   // [vc][n]
    __shared__ float cmaxs[DH], cinvs[DH];

    const int h = blockIdx.x;
    const int slice = blockIdx.y;
    const int tid = threadIdx.x;
    const int lane = tid & 31, wid = tid >> 5;
    const int lm = lane & 15, lh = lane >> 4;
    const int kcs = (wid & 3) * 16;   // kc strip owned by this wave
    const int par = wid >> 2;         // n-chunk parity

    if (tid < DH) {
        cmaxs[tid] = dec_f32(colmax_u[h * DH + tid]);
        cinvs[tid] = 1.0f / colsum[h * DH + tid];
    }
    __syncthreads();

    v8f acc[4];
    #pragma unroll
    for (int c = 0; c < 4; ++c)
        #pragma unroll
        for (int e = 0; e < 8; ++e) acc[c][e] = 0.0f;

    const int colg = tid & 63;
    const int rsub = tid >> 6;

    for (int ch = 0; ch < 8; ++ch) {
        const int nbase = (slice * 8 + ch) * NT;
        for (int rp = 0; rp < NT; rp += 4) {
            const int nl = rp + rsub;
            const size_t g = (size_t)(nbase + nl) * QKV;
            const float kf = (float)kqv[g + h * DH + colg];
            klds[colg * NTP + nl] = (bf16)(__expf(kf - cmaxs[colg]) * cinvs[colg]);
            vlds[colg * NTP + nl] = kqv[g + 2 * K_DIM + h * DH + colg];
        }
        __syncthreads();

        for (int cb = par; cb < NT / 32; cb += 2) {
            const int nb = cb * 32;
            const bf16* pa = &klds[(kcs + lm) * NTP + nb + lh * 8];
            v16bf af = load_frag(pa, pa + 16);
            #pragma unroll
            for (int vb = 0; vb < 4; ++vb) {
                const bf16* pb = &vlds[(vb * 16 + lm) * NTP + nb + lh * 8];
                v16bf bf_ = load_frag(pb, pb + 16);
                acc[vb] = wmma_bf16(af, bf_, acc[vb]);
            }
        }
        __syncthreads();
    }

    #pragma unroll
    for (int vb = 0; vb < 4; ++vb) {
        const int vc = vb * 16 + lm;
        #pragma unroll
        for (int r = 0; r < 8; ++r) {
            const int kc = kcs + r + lh * 8;
            atomicAdd(&ctx[h * (DH * DH) + kc * DH + vc], acc[vb][r]);
        }
    }
}

// ---------------- att[n, h*64+vc] = sum_kc q[n,h,kc] * ctx[h,kc,vc] ----------------
// q tile (128x64 bf16) staged by TDM with padded LDS rows; ctx staged transposed manually.

#define QP 72

__global__ __launch_bounds__(256) void attend_kernel(const bf16* __restrict__ kqv,
                                                     const float* __restrict__ ctx,
                                                     bf16* __restrict__ att) {
    __shared__ alignas(16) bf16 qlds[128 * QP];   // [n][kc]
    __shared__ alignas(16) bf16 ctxT[DH * QP];    // [vc][kc]

    const int h = blockIdx.x;
    const int nb0 = blockIdx.y * 128;
    const int tid = threadIdx.x;
    const int wid = tid >> 5;

    // TDM: q tile 128 rows x 64 bf16 = 128B rows + 16B pad -> LDS stride 144B (QP=72)
    if (wid == 0)
        tdm_load_2d_bf16(lds_off(&qlds[0]),
                         kqv + (size_t)nb0 * QKV + K_DIM + h * DH,
                         DH, 128, QKV, QKV, N_TOK,
                         4u /*32 DW = 128B*/, 3u /*4 DW = 16B*/);

    for (int i = tid; i < DH * DH; i += 256) {
        const int kc = i >> 6, vc = i & 63;
        ctxT[vc * QP + kc] = (bf16)ctx[h * (DH * DH) + i];
    }
    if (wid == 0) __builtin_amdgcn_s_wait_tensorcnt(0);
    __syncthreads();

    const int lane = tid & 31;
    const int lm = lane & 15, lh = lane >> 4;
    const int wr = wid & 3;   // 4 groups x 32 tokens
    const int wc = wid >> 2;  // 2 groups x 32 vc

    v8f acc[2][2];
    #pragma unroll
    for (int i = 0; i < 2; ++i)
        #pragma unroll
        for (int c = 0; c < 2; ++c)
            #pragma unroll
            for (int e = 0; e < 8; ++e) acc[i][c][e] = 0.0f;

    #pragma unroll
    for (int kb = 0; kb < DH; kb += 32) {
        v16bf af[2];
        #pragma unroll
        for (int i = 0; i < 2; ++i) {
            const bf16* p = &qlds[(wr * 32 + i * 16 + lm) * QP + kb + lh * 8];
            af[i] = load_frag(p, p + 16);
        }
        #pragma unroll
        for (int c = 0; c < 2; ++c) {
            const bf16* p = &ctxT[(wc * 32 + c * 16 + lm) * QP + kb + lh * 8];
            v16bf bf_ = load_frag(p, p + 16);
            #pragma unroll
            for (int i = 0; i < 2; ++i) acc[i][c] = wmma_bf16(af[i], bf_, acc[i][c]);
        }
    }

    #pragma unroll
    for (int i = 0; i < 2; ++i)
        #pragma unroll
        for (int c = 0; c < 2; ++c)
            #pragma unroll
            for (int r = 0; r < 8; ++r) {
                const int n = nb0 + wr * 32 + i * 16 + r + lh * 8;
                const int vc = wc * 32 + c * 16 + lm;
                att[(size_t)n * K_DIM + h * DH + vc] = (bf16)acc[i][c][r];
            }
}

// ---------------- host launch ----------------

extern "C" void kernel_launch(void* const* d_in, const int* in_sizes, int n_in,
                              void* d_out, int out_size, void* d_ws, size_t ws_size,
                              hipStream_t stream) {
    (void)in_sizes; (void)n_in; (void)out_size; (void)ws_size;

    const float* input_ = (const float*)d_in[0];
    const float* Wk = (const float*)d_in[1];
    const float* bk = (const float*)d_in[2];
    const float* Wq = (const float*)d_in[3];
    const float* bq = (const float*)d_in[4];
    const float* Wv = (const float*)d_in[5];
    const float* bv = (const float*)d_in[6];
    const float* Wr = (const float*)d_in[7];
    const float* br = (const float*)d_in[8];

    char* ws = (char*)d_ws;
    size_t off = 0;
    auto alloc = [&](size_t bytes) { size_t o = off; off = (off + bytes + 255) & ~(size_t)255; return o; };

    bf16*     inb      = (bf16*)    (ws + alloc((size_t)N_TOK * C_DIM * 2));   // 128 MB
    bf16*     Wqkv     = (bf16*)    (ws + alloc((size_t)C_DIM * QKV * 2));     // 6 MB
    float*    bqkv     = (float*)   (ws + alloc((size_t)QKV * 4));
    bf16*     Wrb      = (bf16*)    (ws + alloc((size_t)K_DIM * K_DIM * 2));   // 2 MB
    bf16*     kqv      = (bf16*)    (ws + alloc((size_t)N_TOK * QKV * 2));     // 384 MB
    bf16*     attb     = (bf16*)    (ws + alloc((size_t)N_TOK * K_DIM * 2));   // 128 MB
    unsigned* colmax_u = (unsigned*)(ws + alloc((size_t)K_DIM * 4));
    float*    colsum   = (float*)   (ws + alloc((size_t)K_DIM * 4));
    float*    ctxf     = (float*)   (ws + alloc((size_t)H_CNT * DH * DH * 4)); // 256 KB

    // 1. convert / pack
    cvt_f32_bf16_kernel<<<4096, 256, 0, stream>>>(input_, inb, N_TOK * C_DIM);
    packw_kernel<<<3072, 256, 0, stream>>>(Wk, Wq, Wv, Wqkv);
    packb_kernel<<<12, 256, 0, stream>>>(bk, bq, bv, bqkv);
    cvt_f32_bf16_kernel<<<1024, 256, 0, stream>>>(Wr, Wrb, K_DIM * K_DIM);
    init_stats_kernel<<<256, 256, 0, stream>>>(colmax_u, colsum, ctxf);

    // 2. fused QKV GEMM: [65536x1024] x [1024x3072] + bias -> kqv (bf16)
    gemm_bf16_kernel<true><<<dim3(QKV / BN, N_TOK / BM), 256, 0, stream>>>(
        inb, Wqkv, bqkv, (void*)kqv, N_TOK, QKV, C_DIM);

    // 3. key column softmax stats (over tokens)
    colmax_kernel<<<dim3(4, 128), 256, 0, stream>>>(kqv, colmax_u);
    colsum_kernel<<<dim3(4, 128), 256, 0, stream>>>(kqv, colmax_u, colsum);

    // 4. query softmax over 64 head features
    qsoftmax_kernel<<<(N_TOK * H_CNT) / 8, 256, 0, stream>>>(kqv);

    // 5. ctx = softmax(k)^T @ v per head
    ctx_kernel<<<dim3(H_CNT, 64), 256, 0, stream>>>(kqv, colmax_u, colsum, ctxf);

    // 6. att = q @ ctx per head
    attend_kernel<<<dim3(H_CNT, N_TOK / 128), 256, 0, stream>>>(kqv, ctxf, attb);

    // 7. out = att @ Wr + br  (fp32 output)
    gemm_bf16_kernel<false><<<dim3(K_DIM / BN, N_TOK / BM), 256, 0, stream>>>(
        attb, Wrb, br, d_out, N_TOK, K_DIM, C_DIM);
}
